// Axial3DWithoutPositionBlock_51316269252946
// MI455X (gfx1250) — compile-verified, matrix operand order checked by reading, not run on hardware
//
#include <hip/hip_runtime.h>
#include <hip/hip_bf16.h>
#include <stdint.h>

typedef __attribute__((ext_vector_type(16))) __bf16 v16bf;
typedef __attribute__((ext_vector_type(8)))  float  v8f;
typedef __attribute__((ext_vector_type(2)))  float  v2f;

#define EPSV 1e-5f

// ---- CDNA5 async global->LDS staging (ASYNCcnt-tracked, bypasses VGPRs) ----
// Generic pointers to LDS carry the LDS byte offset in their low 32 bits
// (LDS aperture: addr[31:0] = LDS address), which is what the VDST operand of
// global_load_async_to_lds_* expects.
__device__ __forceinline__ unsigned lds_off(const void* p) {
  return (unsigned)(uintptr_t)p;
}
__device__ __forceinline__ void async_ld_b32(void* dst, const void* src) {
  asm volatile("global_load_async_to_lds_b32 %0, %1, off"
               :: "v"(lds_off(dst)), "v"((unsigned long long)(uintptr_t)src)
               : "memory");
}
__device__ __forceinline__ void async_ld_b128(void* dst, const void* src) {
  asm volatile("global_load_async_to_lds_b128 %0, %1, off"
               :: "v"(lds_off(dst)), "v"((unsigned long long)(uintptr_t)src)
               : "memory");
}
__device__ __forceinline__ void async_wait() {
  asm volatile("s_wait_asynccnt 0x0" ::: "memory");
}

// K index for element e (0..15) of a 16-bit A/B WMMA fragment (16x16x32),
// per CDNA5 ISA 7.12.2: VGPR r<4 -> K = khalf*8 + 2r + lo/hi; r>=4 -> +16.
__device__ __forceinline__ int wk16(int e, int khalf) {
  return ((e >> 3) << 4) + khalf * 8 + (((e >> 1) & 3) << 1) + (e & 1);
}

// Address into the (N,64,16,56,56) activation buffer for axial direction
// gathers/scatters. b enumerates the batch axis of the flattened (B,C,L)
// view, ax is the attention axis index.
__device__ __forceinline__ int axial_addr(int dir, int b, int c, int ax) {
  if (dir == 0) {            // PERM_IN x: (N,seq,W,C,H) -> axis = H
    int n = b / 896, s = (b / 56) % 16, w = b % 56;
    return (((n * 64 + c) * 16 + s) * 56 + ax) * 56 + w;
  } else if (dir == 1) {     // PERM_IN y: (N,seq,H,C,W) -> axis = W
    int n = b / 896, s = (b / 56) % 16, h = b % 56;
    return (((n * 64 + c) * 16 + s) * 56 + h) * 56 + ax;
  } else {                   // PERM_IN seq: (N,H,W,C,seq) -> axis = seq
    int n = b / 3136, h = (b / 56) % 56, w = b % 56;
    return (((n * 64 + c) * 16 + ax) * 56 + h) * 56 + w;
  }
}

// ---------------- down projection: out[n,o,p] = sum_c cd_w[o,c] x[n,c,p] ----
// O=64, Cin=128, J = 2*50176 = 100352. Block: 8 waves = 4 Mtiles x 2 Ntiles.
// Double-buffered LDS staging via async-to-LDS.
__global__ void k_gemm_down(const float* __restrict__ x, const float* __restrict__ w,
                            float* __restrict__ out) {
  __shared__ __align__(16) float sW[2][64 * 32];
  __shared__ __align__(16) float sX[2][32 * 32];
  const int t = threadIdx.x;
  const int lane = t & 31, wid = t >> 5;
  const int mt = wid & 3, nt = wid >> 2;
  const int jBase = blockIdx.x * 32;            // 50176 % 32 == 0: no n-straddle
  v8f acc = {};

  auto stage = [&](int ks, int bi) {
    const int k0 = ks * 32;
    for (int qd = t; qd < 512; qd += 256) {     // W tile: 64x32, B128 quads
      int o = (qd * 4) >> 5, kk = (qd * 4) & 31;
      async_ld_b128(&sW[bi][qd * 4], &w[o * 128 + k0 + kk]);
    }
    {                                           // X tile: 32x32, 256 quads
      int c = (t * 4) >> 5, jj = (t * 4) & 31;
      int j = jBase + jj;
      int n = j / 50176, p = j % 50176;
      async_ld_b128(&sX[bi][t * 4], &x[(n * 128 + k0 + c) * 50176 + p]);
    }
  };

  stage(0, 0);
  async_wait();
  __syncthreads();
  for (int ks = 0; ks < 4; ++ks) {
    const int bi = ks & 1;
    if (ks + 1 < 4) stage(ks + 1, bi ^ 1);      // overlap next staging w/ WMMA
    const int khalf = lane >> 4, ml = lane & 15;
    v16bf a, b;
#pragma unroll
    for (int e = 0; e < 16; ++e) {
      int k = wk16(e, khalf);
      a[e] = (__bf16)sW[bi][(mt * 16 + ml) * 32 + k];
      b[e] = (__bf16)sX[bi][k * 32 + nt * 16 + ml];
    }
    acc = __builtin_amdgcn_wmma_f32_16x16x32_bf16(false, a, false, b, (short)0, acc, false, false);
    if (ks + 1 < 4) { async_wait(); __syncthreads(); }
  }
  const int nl = lane & 15;
  const int j = jBase + nt * 16 + nl;
  const int n = j / 50176, p = j % 50176;
#pragma unroll
  for (int r = 0; r < 8; ++r) {
    int o = mt * 16 + r + ((lane >> 4) << 3);
    out[(n * 64 + o) * 50176 + p] = acc[r];
  }
}

// ---------------- qkv projection: qkv[b,o,ax] = sum_c w[o,c] cur[gather] ----
// O=128, Cin=64, J = B*L = 100352. Block: 8 waves = 8 Mtiles x 1 Ntile.
__global__ void k_gemm_qkv(const float* __restrict__ cur, const float* __restrict__ w,
                           float* __restrict__ qkv, int dir, int L) {
  __shared__ __align__(16) float sW[2][128 * 32];
  __shared__ __align__(16) float sX[2][32 * 16];
  const int t = threadIdx.x;
  const int lane = t & 31, wid = t >> 5;
  const int jBase = blockIdx.x * 16;
  v8f acc = {};

  auto stage = [&](int ks, int bi) {
    const int k0 = ks * 32;
    for (int qd = t; qd < 1024; qd += 256) {    // W tile: 128x32 B128 quads
      int o = (qd * 4) >> 5, kk = (qd * 4) & 31;
      async_ld_b128(&sW[bi][qd * 4], &w[o * 64 + k0 + kk]);
    }
    for (int i = t; i < 512; i += 256) {        // X tile: strided gather, B32
      int c = i >> 4, jj = i & 15;
      int j = jBase + jj;
      int b = j / L, ax = j % L;
      async_ld_b32(&sX[bi][i], &cur[axial_addr(dir, b, k0 + c, ax)]);
    }
  };

  stage(0, 0);
  async_wait();
  __syncthreads();
  for (int ks = 0; ks < 2; ++ks) {
    const int bi = ks & 1;
    if (ks + 1 < 2) stage(ks + 1, bi ^ 1);
    const int khalf = lane >> 4, ml = lane & 15;
    v16bf a, bb;
#pragma unroll
    for (int e = 0; e < 16; ++e) {
      int k = wk16(e, khalf);
      a[e] = (__bf16)sW[bi][(wid * 16 + ml) * 32 + k];
      bb[e] = (__bf16)sX[bi][k * 16 + ml];
    }
    acc = __builtin_amdgcn_wmma_f32_16x16x32_bf16(false, a, false, bb, (short)0, acc, false, false);
    if (ks + 1 < 2) { async_wait(); __syncthreads(); }
  }
  const int nl = lane & 15;
  const int j = jBase + nl;
  const int b = j / L, ax = j % L;
#pragma unroll
  for (int r = 0; r < 8; ++r) {
    int o = wid * 16 + r + ((lane >> 4) << 3);
    qkv[(b * 128 + o) * L + ax] = acc[r];
  }
}

// ---------------- up projection: out2[n,o,p] = sum_c cu_w[o,c] relu(cur) ----
__global__ void k_gemm_up(const float* __restrict__ cur, const float* __restrict__ w,
                          float* __restrict__ out2) {
  __shared__ __align__(16) float sW[2][128 * 32];
  __shared__ __align__(16) float sX[2][32 * 16];
  const int t = threadIdx.x;
  const int lane = t & 31, wid = t >> 5;
  const int jBase = blockIdx.x * 16;            // 50176 % 16 == 0
  v8f acc = {};

  auto stage = [&](int ks, int bi) {
    const int k0 = ks * 32;
    for (int qd = t; qd < 1024; qd += 256) {
      int o = (qd * 4) >> 5, kk = (qd * 4) & 31;
      async_ld_b128(&sW[bi][qd * 4], &w[o * 64 + k0 + kk]);
    }
    for (int qd = t; qd < 128; qd += 256) {     // X tile: 32x16, contiguous
      int c = (qd * 4) >> 4, jj = (qd * 4) & 15;
      int j = jBase + jj;
      int n = j / 50176, p = j % 50176;
      async_ld_b128(&sX[bi][qd * 4], &cur[(n * 64 + k0 + c) * 50176 + p]);
    }
  };

  stage(0, 0);
  async_wait();
  __syncthreads();
  for (int ks = 0; ks < 2; ++ks) {
    const int bi = ks & 1;
    if (ks + 1 < 2) stage(ks + 1, bi ^ 1);
    const int khalf = lane >> 4, ml = lane & 15;
    v16bf a, bb;
#pragma unroll
    for (int e = 0; e < 16; ++e) {
      int k = wk16(e, khalf);
      a[e] = (__bf16)sW[bi][(wid * 16 + ml) * 32 + k];
      bb[e] = (__bf16)fmaxf(sX[bi][k * 16 + ml], 0.f);   // fused ReLU
    }
    acc = __builtin_amdgcn_wmma_f32_16x16x32_bf16(false, a, false, bb, (short)0, acc, false, false);
    if (ks + 1 < 2) { async_wait(); __syncthreads(); }
  }
  const int nl = lane & 15;
  const int j = jBase + nl;
  const int n = j / 50176, p = j % 50176;
#pragma unroll
  for (int r = 0; r < 8; ++r) {
    int o = wid * 16 + r + ((lane >> 4) << 3);
    out2[(n * 128 + o) * 50176 + p] = acc[r];
  }
}

// ---------------- qk[i,j] = sum_c q[c,i] k[c,j] (K=4, fp32 WMMA) ------------
__global__ void k_qk(const float* __restrict__ qkv, float* __restrict__ qk, int L) {
  const int bg = blockIdx.x;
  const int b = bg >> 3, g = bg & 7;
  __shared__ __align__(16) float sq[4 * 56];
  __shared__ __align__(16) float sk[4 * 56];
  const int t = threadIdx.x;
  const int base = (b * 128 + g * 16) * L;      // 4*L contiguous rows; L%4==0
  for (int qd = t; qd < L; qd += 256) async_ld_b128(&sq[qd * 4], &qkv[base + qd * 4]);
  for (int qd = t; qd < L; qd += 256) async_ld_b128(&sk[qd * 4], &qkv[base + 4 * L + qd * 4]);
  async_wait();
  __syncthreads();
  const int lane = t & 31, wid = t >> 5;
  const int khalf = lane >> 4, nl = lane & 15;
  const int nT = (L + 15) >> 4;
  float* qko = qk + (long long)bg * L * L;
  for (int tile = wid; tile < nT * nT; tile += 8) {
    const int ti = tile / nT, tj = tile % nT;
    v2f a, bb;
#pragma unroll
    for (int r = 0; r < 2; ++r) {
      int c = khalf * 2 + r;               // K index, per 32-bit A 16x4 layout
      int i = ti * 16 + nl;
      int j = tj * 16 + nl;
      a[r]  = (i < L) ? sq[c * L + i] : 0.f;
      bb[r] = (j < L) ? sk[c * L + j] : 0.f;
    }
    v8f acc = {};
    acc = __builtin_amdgcn_wmma_f32_16x16x4_f32(false, a, false, bb, (short)0, acc, false, false);
#pragma unroll
    for (int r = 0; r < 8; ++r) {
      int i = ti * 16 + r + (khalf << 3);
      int j = tj * 16 + nl;
      if (i < L && j < L) qko[i * L + j] = acc[r];
    }
  }
}

// ------- fused qk-BN + softmax + sv[c,i] = sum_j sim[i,j] v[c,j] (bf16) -----
__global__ void k_attn_sv(const float* __restrict__ qk, const float* __restrict__ qkv,
                          const float* __restrict__ scale, const float* __restrict__ shift,
                          float* __restrict__ sv, int L) {
  const int bg = blockIdx.x;
  const int b = bg >> 3, g = bg & 7;
  __shared__ float sim[56 * 56];
  __shared__ __align__(16) float vld[8 * 56];
  const int t = threadIdx.x;
  const float sg = scale[g], sb = shift[g];
  const float* qki = qk + (long long)bg * L * L;
  const int vbase = (b * 128 + g * 16 + 8) * L;
  for (int qd = t; qd < 2 * L; qd += 256) async_ld_b128(&vld[qd * 4], &qkv[vbase + qd * 4]);
  for (int i = t; i < L * L; i += 256) sim[i] = sg * qki[i] + sb;   // BN fused
  async_wait();
  __syncthreads();
  if (t < L) {
    float mx = -3.4e38f;
    for (int j = 0; j < L; ++j) mx = fmaxf(mx, sim[t * L + j]);
    float sum = 0.f;
    for (int j = 0; j < L; ++j) { float e = __expf(sim[t * L + j] - mx); sim[t * L + j] = e; sum += e; }
    float inv = 1.f / sum;
    for (int j = 0; j < L; ++j) sim[t * L + j] *= inv;
  }
  __syncthreads();
  const int lane = t & 31, wid = t >> 5;
  const int nT = (L + 15) >> 4;
  const int kSteps = (L + 31) >> 5;
  if (wid < nT) {
    const int khalf = lane >> 4, nl = lane & 15;
    const int ti = wid;
    v8f acc = {};
    for (int ks = 0; ks < kSteps; ++ks) {
      v16bf a, bb;
#pragma unroll
      for (int e = 0; e < 16; ++e) {
        int j = ks * 32 + wk16(e, khalf);
        int i = ti * 16 + nl;
        a[e]  = (nl < 8 && j < L) ? (__bf16)vld[nl * L + j] : (__bf16)0.f;
        bb[e] = (i < L && j < L) ? (__bf16)sim[i * L + j] : (__bf16)0.f;
      }
      acc = __builtin_amdgcn_wmma_f32_16x16x32_bf16(false, a, false, bb, (short)0, acc, false, false);
    }
    if (lane < 16) {                        // D rows M=0..7 live in lanes 0-15
#pragma unroll
      for (int r = 0; r < 8; ++r) {
        int i = ti * 16 + nl;
        if (i < L) sv[(b * 64 + g * 8 + r) * L + i] = acc[r];
      }
    }
  }
}

// ---------------- BN statistics / finalize / apply --------------------------
__global__ void k_stats(const float* __restrict__ buf, float* __restrict__ acc,
                        long long total, int C, long long S) {
  __shared__ float sh[256];
  for (int i = threadIdx.x; i < 2 * C; i += blockDim.x) sh[i] = 0.f;
  __syncthreads();
  const long long stride = (long long)gridDim.x * blockDim.x;
  for (long long idx = (long long)blockIdx.x * blockDim.x + threadIdx.x; idx < total; idx += stride) {
    float v = buf[idx];
    int ch = (int)((idx / S) % C);
    atomicAdd(&sh[ch], v);
    atomicAdd(&sh[C + ch], v * v);
  }
  __syncthreads();
  for (int i = threadIdx.x; i < 2 * C; i += blockDim.x) atomicAdd(&acc[i], sh[i]);
}

__global__ void k_finalize(const float* __restrict__ acc, const float* __restrict__ gamma,
                           const float* __restrict__ beta, float* __restrict__ scale,
                           float* __restrict__ shift, int C, float cinv) {
  int t = blockIdx.x * blockDim.x + threadIdx.x;
  if (t < C) {
    float mean = acc[t] * cinv;
    float var  = acc[C + t] * cinv - mean * mean;
    float s = gamma[t] * rsqrtf(var + EPSV);
    scale[t] = s;
    shift[t] = beta[t] - mean * s;
  }
}

__global__ void k_apply(float* __restrict__ buf, const float* __restrict__ scale,
                        const float* __restrict__ shift, long long total, int C,
                        long long S, int relu) {
  const long long stride = (long long)gridDim.x * blockDim.x;
  for (long long idx = (long long)blockIdx.x * blockDim.x + threadIdx.x; idx < total; idx += stride) {
    if (idx + stride < total) __builtin_prefetch(&buf[idx + stride], 0, 1);
    int ch = (int)((idx / S) % C);
    float v = scale[ch] * buf[idx] + shift[ch];
    if (relu) v = fmaxf(v, 0.f);
    buf[idx] = v;
  }
}

// sv BN-apply + inverse-permute scatter back into (N,64,16,56,56)
__global__ void k_sv_scatter(const float* __restrict__ sv, const float* __restrict__ scale,
                             const float* __restrict__ shift, float* __restrict__ outn,
                             int dir, int L, long long total) {
  const long long stride = (long long)gridDim.x * blockDim.x;
  for (long long idx = (long long)blockIdx.x * blockDim.x + threadIdx.x; idx < total; idx += stride) {
    int ax = (int)(idx % L);
    int c  = (int)((idx / L) & 63);
    int b  = (int)(idx / (64LL * L));
    float v = scale[c] * sv[idx] + shift[c];
    outn[axial_addr(dir, b, c, ax)] = v;
  }
}

// final: relu(bn2(out2) + identity)
__global__ void k_final(const float* __restrict__ out2, const float* __restrict__ scale,
                        const float* __restrict__ shift, const float* __restrict__ x,
                        float* __restrict__ out, long long total) {
  const long long stride = (long long)gridDim.x * blockDim.x;
  for (long long idx = (long long)blockIdx.x * blockDim.x + threadIdx.x; idx < total; idx += stride) {
    int ch = (int)((idx / 50176) % 128);
    float v = scale[ch] * out2[idx] + shift[ch] + x[idx];
    out[idx] = fmaxf(v, 0.f);
  }
}

extern "C" void kernel_launch(void* const* d_in, const int* in_sizes, int n_in,
                              void* d_out, int out_size, void* d_ws, size_t ws_size,
                              hipStream_t stream) {
  const float* x     = (const float*)d_in[0];
  const float* cd_w  = (const float*)d_in[1];
  const float* bn1_g = (const float*)d_in[2];
  const float* bn1_b = (const float*)d_in[3];
  const float* qkv_w = (const float*)d_in[4];   // (3,128,64)
  const float* bnq_g = (const float*)d_in[5];
  const float* bnq_b = (const float*)d_in[6];
  const float* bns_g = (const float*)d_in[7];
  const float* bns_b = (const float*)d_in[8];
  const float* bno_g = (const float*)d_in[9];
  const float* bno_b = (const float*)d_in[10];
  const float* cu_w  = (const float*)d_in[11];  // (128,64)
  const float* bn2_g = (const float*)d_in[12];
  const float* bn2_b = (const float*)d_in[13];
  float* out = (float*)d_out;

  float* ws = (float*)d_ws;
  const long long SZ1   = 6422528LL;    // (2,64,50176)
  const long long SZQKV = 12845056LL;   // (B,128,L) == (2,128,50176)
  const long long SZQK  = 44957696LL;   // max (B,8,L,L)
  float* bufA = ws;
  float* bufB = bufA + SZ1;
  float* qkvB = bufB + SZ1;             // reused as out2 after direction loop
  float* qkB  = qkvB + SZQKV;
  float* svB  = qkB + SZQK;
  float* accB = svB + SZ1;
  float* scB  = accB + 256;
  float* shB  = scB + 128;

  auto stats = [&](const float* buf, long long total, int C, long long S,
                   const float* g, const float* bta, float cinv) {
    hipMemsetAsync(accB, 0, 2 * C * sizeof(float), stream);
    k_stats<<<1024, 256, 0, stream>>>(buf, accB, total, C, S);
    k_finalize<<<1, 128, 0, stream>>>(accB, g, bta, scB, shB, C, cinv);
  };

  // down projection, bn1, relu
  k_gemm_down<<<3136, 256, 0, stream>>>(x, cd_w, bufA);
  stats(bufA, SZ1, 64, 50176LL, bn1_g, bn1_b, 1.f / 100352.f);
  k_apply<<<2048, 256, 0, stream>>>(bufA, scB, shB, SZ1, 64, 50176LL, 1);

  float* cur = bufA;
  float* nxt = bufB;
  for (int d = 0; d < 3; ++d) {
    const int L = (d == 2) ? 16 : 56;
    const long long B = (d == 2) ? 6272 : 1792;
    k_gemm_qkv<<<6272, 256, 0, stream>>>(cur, qkv_w + d * 128 * 64, qkvB, d, L);
    stats(qkvB, SZQKV, 128, (long long)L, bnq_g + d * 128, bnq_b + d * 128, 1.f / 100352.f);
    k_apply<<<2048, 256, 0, stream>>>(qkvB, scB, shB, SZQKV, 128, (long long)L, 0);
    k_qk<<<(int)(B * 8), 256, 0, stream>>>(qkvB, qkB, L);
    const long long qkTot = B * 8 * (long long)L * L;
    stats(qkB, qkTot, 8, (long long)L * L, bns_g + d * 8, bns_b + d * 8,
          1.f / (float)(B * (long long)L * L));
    k_attn_sv<<<(int)(B * 8), 256, 0, stream>>>(qkB, qkvB, scB, shB, svB, L);
    stats(svB, SZ1, 64, (long long)L, bno_g + d * 64, bno_b + d * 64, 1.f / 100352.f);
    k_sv_scatter<<<2048, 256, 0, stream>>>(svB, scB, shB, nxt, d, L, SZ1);
    float* tmp = cur; cur = nxt; nxt = tmp;
  }

  // relu (fused into gather) + up projection + bn2 + residual + relu
  float* out2 = qkvB;
  k_gemm_up<<<6272, 256, 0, stream>>>(cur, cu_w, out2);
  stats(out2, SZQKV, 128, 50176LL, bn2_g, bn2_b, 1.f / 100352.f);
  k_final<<<2048, 256, 0, stream>>>(out2, scB, shB, x, out, SZQKV);
}
